// HSTUModel_45183055954604
// MI455X (gfx1250) — matrix operand order repaired and checked by reading, not compiled, not run on hardware
//
#include <hip/hip_runtime.h>
#include <hip/hip_bf16.h>

// ---------------------------------------------------------------------------
// HSTU forward for MI455X (gfx1250, wave32). All GEMM-shaped work runs on
// v_wmma_f32_16x16x32_bf16 (bf16 A/B, f32 accum). Elementwise + LN in fp32.
// Global->LDS tile staging uses CDNA5 async-to-LDS DMA (ASYNCcnt), with
// double-buffered LDS tiles so the DMA for tile i+1 overlaps WMMA on tile i.
// ---------------------------------------------------------------------------

#define B_  2
#define S_  2048
#define D_  1024
#define H_  8
#define DA_ 64
#define DL_ 64
#define L_  2
#define M_  (B_ * S_)      // 4096 rows
#define E_  2048           // proj dim = 2*H*DL + 2*H*DA

typedef __attribute__((ext_vector_type(16))) __bf16 v16bf;
typedef __attribute__((ext_vector_type(8)))  __bf16 v8bf;
typedef __attribute__((ext_vector_type(8)))  float  v8f;

__device__ __forceinline__ float silu_f(float x) {
  return x / (1.0f + __expf(-x));
}

__device__ __forceinline__ v8f wmma_bf16(v16bf a, v16bf b, v8f c) {
  return __builtin_amdgcn_wmma_f32_16x16x32_bf16(false, a, false, b,
                                                 (short)0, c, false, false);
}

// CDNA5 async DMA: copy 16 bytes per lane from global directly into LDS.
// Tracked by ASYNCcnt (no VGPR round trip, no loadcnt stall).
// Generic pointers to __shared__ carry the LDS byte offset in their low 32b.
__device__ __forceinline__ void cp_async_b128(void* lds_ptr, const void* gptr) {
  unsigned int lds_off = (unsigned int)(size_t)lds_ptr;
  unsigned long long ga = (unsigned long long)(size_t)gptr;
  asm volatile("global_load_async_to_lds_b128 %0, %1, off"
               :: "v"(lds_off), "v"(ga) : "memory");
}
__device__ __forceinline__ void wait_async0() {
  asm volatile("s_wait_asynccnt 0" ::: "memory");
}

// A fragment: 16x32 (MxK), LDS tile row-major [m][k], row stride ldm elems.
// Lane L: m=L%16, hi=L/16. elems 0..7 -> k=hi*8+j ; elems 8..15 -> k=16+hi*8+j.
__device__ __forceinline__ v16bf load_A_frag(const __bf16* base, int ldm, int lane) {
  int m  = lane & 15;
  int hi = lane >> 4;
  const __bf16* row = base + m * ldm;
  v8bf lo = *(const v8bf*)(row + hi * 8);
  v8bf hh = *(const v8bf*)(row + 16 + hi * 8);
  v16bf r;
#pragma unroll
  for (int j = 0; j < 8; ++j) { r[j] = lo[j]; r[j + 8] = hh[j]; }
  return r;
}

// B fragment: 32x16 (KxN), LDS tile stored TRANSPOSED [n][k], row stride ldk.
// Lane L: n=L%16, kg=L/16. elems 0..7 -> k=kg*8+j ; elems 8..15 -> k=16+kg*8+j.
__device__ __forceinline__ v16bf load_B_frag(const __bf16* baseT, int ldk, int lane) {
  int n  = lane & 15;
  int kg = lane >> 4;
  const __bf16* row = baseT + n * ldk + kg * 8;
  v8bf a = *(const v8bf*)(row);
  v8bf b = *(const v8bf*)(row + 16);
  v16bf r;
#pragma unroll
  for (int j = 0; j < 8; ++j) { r[j] = a[j]; r[j + 8] = b[j]; }
  return r;
}

// ---------------------------------------------------------------------------
// fp32 -> bf16 weight conversion
__global__ void cvt_bf16_kernel(const float* __restrict__ in,
                                __bf16* __restrict__ out, int n) {
  int i = blockIdx.x * blockDim.x + threadIdx.x;
  if (i < n) out[i] = (__bf16)in[i];
}

// ---------------------------------------------------------------------------
// LayerNorm over D=1024; 256 threads/block (8 waves), one row per block.
template <bool BF16OUT>
__global__ void ln_kernel(const float* __restrict__ x, const float* __restrict__ w,
                          const float* __restrict__ b, void* __restrict__ out,
                          float eps) {
  __shared__ float red_s[8], red_q[8];
  const int row = blockIdx.x;
  const int tid = threadIdx.x;
  const float* xr = x + (size_t)row * D_;
  float s = 0.f, q = 0.f;
#pragma unroll
  for (int i = tid; i < D_; i += 256) { float v = xr[i]; s += v; q += v * v; }
#pragma unroll
  for (int o = 16; o > 0; o >>= 1) {
    s += __shfl_xor(s, o, 32);
    q += __shfl_xor(q, o, 32);
  }
  if ((tid & 31) == 0) { red_s[tid >> 5] = s; red_q[tid >> 5] = q; }
  __syncthreads();
  if (tid == 0) {
    float ts = 0.f, tq = 0.f;
#pragma unroll
    for (int i = 0; i < 8; ++i) { ts += red_s[i]; tq += red_q[i]; }
    red_s[0] = ts; red_q[0] = tq;
  }
  __syncthreads();
  const float mu  = red_s[0] * (1.0f / D_);
  const float var = red_q[0] * (1.0f / D_) - mu * mu;
  const float rs  = rsqrtf(var + eps);
#pragma unroll
  for (int i = tid; i < D_; i += 256) {
    float v = (xr[i] - mu) * rs * w[i] + b[i];
    if (BF16OUT) ((__bf16*)out)[(size_t)row * D_ + i] = (__bf16)v;
    else         ((float*)out)[(size_t)row * D_ + i]  = v;
  }
}

// ---------------------------------------------------------------------------
// proj GEMM: C(M x E) = silu(h(M x D) @ W(D x E)); scatter u/v/q/k.
// 128 threads (4 waves); tile 64x64, K-step 32. Double-buffered:
// A tile k+1 via async DMA, W tile k+1 staged in VGPRs, both while WMMA runs.
__global__ void proj_gemm_kernel(const __bf16* __restrict__ A,   // M x D
                                 const __bf16* __restrict__ W,   // D x E
                                 float*  __restrict__ u_out,     // M x 512
                                 __bf16* __restrict__ v_out,     // (B*H,S,64)
                                 __bf16* __restrict__ q_out,
                                 __bf16* __restrict__ k_out) {
  __shared__ __bf16 aT[2][64 * 32];  // [m][k]
  __shared__ __bf16 bT[2][64 * 32];  // [n][k]
  const int tid  = threadIdx.x;
  const int lane = tid & 31;
  const int w    = tid >> 5;
  const int m0   = blockIdx.y * 64;
  const int n0   = blockIdx.x * 64;
  const int NSTEP = D_ / 32;
  v8f acc[4] = {};
  // ---- prologue: stage tile 0
#pragma unroll
  for (int it = 0; it < 2; ++it) {
    int idx = tid + it * 128;
    int r = idx >> 2, c8 = idx & 3;
    cp_async_b128(&aT[0][r * 32 + c8 * 8], &A[(size_t)(m0 + r) * D_ + c8 * 8]);
  }
#pragma unroll
  for (int it = 0; it < 2; ++it) {
    int idx = tid + it * 128;
    int kk = idx >> 3, c8 = idx & 7;
    v8bf d = *(const v8bf*)&W[(size_t)kk * E_ + n0 + c8 * 8];
#pragma unroll
    for (int j = 0; j < 8; ++j) bT[0][(c8 * 8 + j) * 32 + kk] = d[j];
  }
  wait_async0();
  __syncthreads();
  // ---- main pipeline
  v8bf breg[2];
  for (int step = 0; step < NSTEP; ++step) {
    const int cur = step & 1, nxt = cur ^ 1;
    const int k1  = (step + 1) * 32;
    const bool pf = (step + 1 < NSTEP);
    if (pf) {
#pragma unroll
      for (int it = 0; it < 2; ++it) {          // A tile k+1: async DMA
        int idx = tid + it * 128;
        int r = idx >> 2, c8 = idx & 3;
        cp_async_b128(&aT[nxt][r * 32 + c8 * 8],
                      &A[(size_t)(m0 + r) * D_ + k1 + c8 * 8]);
      }
#pragma unroll
      for (int it = 0; it < 2; ++it) {          // W tile k+1: stage in VGPRs
        int idx = tid + it * 128;
        int kk = idx >> 3, c8 = idx & 7;
        breg[it] = *(const v8bf*)&W[(size_t)(k1 + kk) * E_ + n0 + c8 * 8];
      }
    }
    // compute tile k (global prefetches in flight)
    v16bf af = load_A_frag(&aT[cur][w * 16 * 32], 32, lane);
#pragma unroll
    for (int nt = 0; nt < 4; ++nt) {
      v16bf bf_ = load_B_frag(&bT[cur][nt * 16 * 32], 32, lane);
      acc[nt] = wmma_bf16(af, bf_, acc[nt]);
    }
    if (pf) {                                   // drain W regs -> LDS transposed
#pragma unroll
      for (int it = 0; it < 2; ++it) {
        int idx = tid + it * 128;
        int kk = idx >> 3, c8 = idx & 7;
#pragma unroll
        for (int j = 0; j < 8; ++j) bT[nxt][(c8 * 8 + j) * 32 + kk] = breg[it][j];
      }
    }
    wait_async0();
    __syncthreads();
  }
  // ---- epilogue
  const int hiL = lane >> 4, nn = lane & 15;
#pragma unroll
  for (int nt = 0; nt < 4; ++nt) {
#pragma unroll
    for (int r = 0; r < 8; ++r) {
      int m = m0 + w * 16 + r + 8 * hiL;
      int e = n0 + nt * 16 + nn;
      float val = silu_f(acc[nt][r]);
      if (e < 512) {
        u_out[(size_t)m * 512 + e] = val;
      } else {
        int seg = e >> 9;          // 1=v, 2=q, 3=k
        int eh  = e & 511;
        int h = eh >> 6, d = eh & 63;
        int bb = m / S_, s = m % S_;
        size_t off = (((size_t)(bb * H_ + h)) * S_ + s) * 64 + d;
        __bf16 bv = (__bf16)val;
        if (seg == 1)      v_out[off] = bv;
        else if (seg == 2) q_out[off] = bv;
        else               k_out[off] = bv;
      }
    }
  }
}

// ---------------------------------------------------------------------------
// Causal SiLU-attention: per (b,h), per 64-query block; stream 64-key blocks.
// out(s,d) += silu(q.k^T)/S (masked) @ v. 128 threads (4 waves).
// K tiles via async-to-LDS DMA, V tiles register-staged + transposed; both
// double-buffered so block tb+1 streams in while block tb computes.
__global__ void attn_kernel(const __bf16* __restrict__ Q, const __bf16* __restrict__ Kk,
                            const __bf16* __restrict__ V, float* __restrict__ O,
                            float inv_seq) {
  __shared__ __bf16 q_lds[64 * 64];        // [s][d]
  __shared__ __bf16 k_lds[2][64 * 64];     // [t][d]  (transposed-B for q.k^T)
  __shared__ __bf16 vT_lds[2][64 * 64];    // [d][t]  (transposed-B for scores.v)
  __shared__ __bf16 s_lds[4][16 * 64];     // per-wave score strip [s][t]
  const int tid  = threadIdx.x;
  const int lane = tid & 31;
  const int w    = tid >> 5;
  const int bh   = blockIdx.y;
  const int sb   = blockIdx.x;
  const int s0   = sb * 64;
  const __bf16* Qb   = Q  + ((size_t)bh * S_ + s0) * 64;
  const __bf16* Kbase = Kk + (size_t)bh * S_ * 64;
  const __bf16* Vbase = V  + (size_t)bh * S_ * 64;
  // ---- prologue: Q tile + key-block 0
#pragma unroll
  for (int it = 0; it < 4; ++it) {
    int idx = tid + it * 128;
    int r = idx >> 3, c8 = idx & 7;
    cp_async_b128(&q_lds[r * 64 + c8 * 8], &Qb[(size_t)r * 64 + c8 * 8]);
    cp_async_b128(&k_lds[0][r * 64 + c8 * 8], &Kbase[(size_t)r * 64 + c8 * 8]);
    v8bf dv = *(const v8bf*)&Vbase[(size_t)r * 64 + c8 * 8];
#pragma unroll
    for (int j = 0; j < 8; ++j) vT_lds[0][(c8 * 8 + j) * 64 + r] = dv[j];
  }
  wait_async0();
  __syncthreads();
  v16bf aq0 = load_A_frag(&q_lds[w * 16 * 64], 64, lane);       // d 0..31
  v16bf aq1 = load_A_frag(&q_lds[w * 16 * 64 + 32], 64, lane);  // d 32..63
  v8f acc[4] = {};
  const int hiL = lane >> 4, nn = lane & 15;
  v8bf vreg[4];
  for (int tb = 0; tb <= sb; ++tb) {
    const int cur = tb & 1, nxt = cur ^ 1;
    const int t0  = tb * 64;
    const bool pf = (tb + 1 <= sb);
    if (pf) {
      const __bf16* Kn = Kbase + (size_t)(t0 + 64) * 64;
      const __bf16* Vn = Vbase + (size_t)(t0 + 64) * 64;
#pragma unroll
      for (int it = 0; it < 4; ++it) {          // K block tb+1: async DMA
        int idx = tid + it * 128;
        int r = idx >> 3, c8 = idx & 7;
        cp_async_b128(&k_lds[nxt][r * 64 + c8 * 8], &Kn[(size_t)r * 64 + c8 * 8]);
        vreg[it] = *(const v8bf*)&Vn[(size_t)r * 64 + c8 * 8]; // V: stage regs
      }
    }
    // ---- compute key-block tb (prefetches in flight)
    __bf16* sw = &s_lds[w][0];
#pragma unroll
    for (int nt = 0; nt < 4; ++nt) {
      v16bf b0 = load_B_frag(&k_lds[cur][nt * 16 * 64], 64, lane);
      v16bf b1 = load_B_frag(&k_lds[cur][nt * 16 * 64 + 32], 64, lane);
      v8f c = {};
      c = wmma_bf16(aq0, b0, c);
      c = wmma_bf16(aq1, b1, c);
#pragma unroll
      for (int r = 0; r < 8; ++r) {
        int srow = s0 + w * 16 + r + 8 * hiL;
        int tcol = t0 + nt * 16 + nn;
        float val = (tcol <= srow) ? silu_f(c[r]) * inv_seq : 0.0f;
        sw[(r + 8 * hiL) * 64 + nt * 16 + nn] = (__bf16)val;
      }
    }
    // scores strip (16x64) @ v (64x64): A from s_lds (same wave, LDS in-order)
    v16bf as0 = load_A_frag(sw, 64, lane);
    v16bf as1 = load_A_frag(sw + 32, 64, lane);
#pragma unroll
    for (int nt = 0; nt < 4; ++nt) {
      v16bf b0 = load_B_frag(&vT_lds[cur][nt * 16 * 64], 64, lane);
      v16bf b1 = load_B_frag(&vT_lds[cur][nt * 16 * 64 + 32], 64, lane);
      acc[nt] = wmma_bf16(as0, b0, acc[nt]);
      acc[nt] = wmma_bf16(as1, b1, acc[nt]);
    }
    if (pf) {                                   // drain V regs -> LDS transposed
#pragma unroll
      for (int it = 0; it < 4; ++it) {
        int idx = tid + it * 128;
        int r = idx >> 3, c8 = idx & 7;
#pragma unroll
        for (int j = 0; j < 8; ++j) vT_lds[nxt][(c8 * 8 + j) * 64 + r] = vreg[it][j];
      }
    }
    wait_async0();
    __syncthreads();
  }
#pragma unroll
  for (int nt = 0; nt < 4; ++nt) {
#pragma unroll
    for (int r = 0; r < 8; ++r) {
      int srow = s0 + w * 16 + r + 8 * hiL;
      O[((size_t)bh * S_ + srow) * 64 + nt * 16 + nn] = acc[nt][r];
    }
  }
}

// ---------------------------------------------------------------------------
// Per-(b,h,s) norm over DL=64 (no affine) then multiply by u -> bf16 u_dot.
// One wave per row; 256 threads = 8 rows/block.
__global__ void norm_udot_kernel(const float* __restrict__ O, const float* __restrict__ U,
                                 __bf16* __restrict__ UD, int nrows) {
  int gw   = (int)((blockIdx.x * blockDim.x + threadIdx.x) >> 5);
  int lane = threadIdx.x & 31;
  if (gw >= nrows) return;
  int bh = gw / S_, s = gw % S_;
  const float* row = O + ((size_t)bh * S_ + s) * 64;
  float v0 = row[lane], v1 = row[lane + 32];
  float sum = v0 + v1, sq = v0 * v0 + v1 * v1;
#pragma unroll
  for (int o = 16; o > 0; o >>= 1) {
    sum += __shfl_xor(sum, o, 32);
    sq  += __shfl_xor(sq, o, 32);
  }
  float mu  = sum * (1.0f / 64.0f);
  float var = sq * (1.0f / 64.0f) - mu * mu;
  float rs  = rsqrtf(var + 1e-6f);
  int bb = bh / H_, h = bh % H_;
  size_t mrow = ((size_t)bb * S_ + s) * 512 + (size_t)h * 64;
  UD[mrow + lane]      = (__bf16)(((v0 - mu) * rs) * U[mrow + lane]);
  UD[mrow + lane + 32] = (__bf16)(((v1 - mu) * rs) * U[mrow + lane + 32]);
}

// ---------------------------------------------------------------------------
// out GEMM: x(M x D) += u_dot(M x 512) @ o_w(D x 512)^T + o_b.
// o_w's natural row-major layout IS the transposed-B [n][k] tile layout,
// so BOTH tiles are straight copies -> both double-buffered async DMA.
__global__ void out_gemm_kernel(const __bf16* __restrict__ A,    // M x 512
                                const __bf16* __restrict__ WoT,  // D x 512 (= [n][k])
                                const float* __restrict__ ob,    // D
                                float* __restrict__ xbuf) {      // M x D in-place
  __shared__ __bf16 aT[2][64 * 32];
  __shared__ __bf16 bT[2][64 * 32];
  const int tid  = threadIdx.x;
  const int lane = tid & 31;
  const int w    = tid >> 5;
  const int m0   = blockIdx.y * 64;
  const int n0   = blockIdx.x * 64;
  const int NSTEP = 512 / 32;
  v8f acc[4] = {};
#pragma unroll
  for (int it = 0; it < 2; ++it) {
    int idx = tid + it * 128;
    int r = idx >> 2, c8 = idx & 3;
    cp_async_b128(&aT[0][r * 32 + c8 * 8], &A[(size_t)(m0 + r) * 512 + c8 * 8]);
    cp_async_b128(&bT[0][r * 32 + c8 * 8], &WoT[(size_t)(n0 + r) * 512 + c8 * 8]);
  }
  wait_async0();
  __syncthreads();
  for (int step = 0; step < NSTEP; ++step) {
    const int cur = step & 1, nxt = cur ^ 1;
    const int k1  = (step + 1) * 32;
    const bool pf = (step + 1 < NSTEP);
    if (pf) {
#pragma unroll
      for (int it = 0; it < 2; ++it) {
        int idx = tid + it * 128;
        int r = idx >> 2, c8 = idx & 3;
        cp_async_b128(&aT[nxt][r * 32 + c8 * 8],
                      &A[(size_t)(m0 + r) * 512 + k1 + c8 * 8]);
        cp_async_b128(&bT[nxt][r * 32 + c8 * 8],
                      &WoT[(size_t)(n0 + r) * 512 + k1 + c8 * 8]);
      }
    }
    v16bf af = load_A_frag(&aT[cur][w * 16 * 32], 32, lane);
#pragma unroll
    for (int nt = 0; nt < 4; ++nt) {
      v16bf bf_ = load_B_frag(&bT[cur][nt * 16 * 32], 32, lane);
      acc[nt] = wmma_bf16(af, bf_, acc[nt]);
    }
    wait_async0();
    __syncthreads();
  }
  const int hiL = lane >> 4, nn = lane & 15;
#pragma unroll
  for (int nt = 0; nt < 4; ++nt) {
#pragma unroll
    for (int r = 0; r < 8; ++r) {
      int m = m0 + w * 16 + r + 8 * hiL;
      int e = n0 + nt * 16 + nn;
      size_t off = (size_t)m * D_ + e;
      xbuf[off] = xbuf[off] + acc[nt][r] + ob[e];
    }
  }
}

// ---------------------------------------------------------------------------
extern "C" void kernel_launch(void* const* d_in, const int* in_sizes, int n_in,
                              void* d_out, int out_size, void* d_ws, size_t ws_size,
                              hipStream_t stream) {
  const float* x_in      = (const float*)d_in[0];
  // d_in[1] = attn_mask (causal tril) -- computed analytically, unused
  const float* uvqk      = (const float*)d_in[2];  // (L, D, 2048)
  const float* o_w       = (const float*)d_in[3];  // (L, D, 512)
  const float* o_b       = (const float*)d_in[4];  // (L, D)
  const float* ln_w      = (const float*)d_in[5];  // (L, D)
  const float* ln_b      = (const float*)d_in[6];  // (L, D)
  const float* last_ln_w = (const float*)d_in[7];  // (D)
  const float* last_ln_b = (const float*)d_in[8];  // (D)
  float* out = (float*)d_out;

  // workspace layout (all sizes 256B-aligned by construction) ~64MB total
  char* p = (char*)d_ws;
  float*  Xbuf = (float*)p;            p += (size_t)M_ * D_ * 4;       // 16MB
  __bf16* Hb   = (__bf16*)p;           p += (size_t)M_ * D_ * 2;       // 8MB
  __bf16* Wp   = (__bf16*)p;           p += (size_t)D_ * E_ * 2;       // 4MB
  __bf16* Wo   = (__bf16*)p;           p += (size_t)D_ * 512 * 2;      // 1MB
  float*  Ub   = (float*)p;            p += (size_t)M_ * 512 * 4;      // 8MB
  __bf16* Qb   = (__bf16*)p;           p += (size_t)B_ * H_ * S_ * 64 * 2; // 4MB
  __bf16* Kb   = (__bf16*)p;           p += (size_t)B_ * H_ * S_ * 64 * 2; // 4MB
  __bf16* Vb   = (__bf16*)p;           p += (size_t)B_ * H_ * S_ * 64 * 2; // 4MB
  float*  Ob   = (float*)p;            p += (size_t)B_ * H_ * S_ * 64 * 4; // 8MB
  __bf16* UD   = (__bf16*)p;           p += (size_t)M_ * 512 * 2;      // 4MB
  (void)ws_size; (void)n_in; (void)in_sizes; (void)out_size;

  hipMemcpyAsync(Xbuf, x_in, (size_t)M_ * D_ * sizeof(float),
                 hipMemcpyDeviceToDevice, stream);

  for (int i = 0; i < L_; ++i) {
    const int nWp = D_ * E_;
    const int nWo = D_ * 512;
    cvt_bf16_kernel<<<(nWp + 255) / 256, 256, 0, stream>>>(
        uvqk + (size_t)i * nWp, Wp, nWp);
    cvt_bf16_kernel<<<(nWo + 255) / 256, 256, 0, stream>>>(
        o_w + (size_t)i * nWo, Wo, nWo);

    ln_kernel<true><<<M_, 256, 0, stream>>>(
        Xbuf, ln_w + (size_t)i * D_, ln_b + (size_t)i * D_, (void*)Hb, 1e-6f);

    proj_gemm_kernel<<<dim3(E_ / 64, M_ / 64), 128, 0, stream>>>(
        Hb, Wp, Ub, Vb, Qb, Kb);

    attn_kernel<<<dim3(S_ / 64, B_ * H_), 128, 0, stream>>>(
        Qb, Kb, Vb, Ob, 1.0f / (float)S_);

    const int nrows = B_ * H_ * S_;
    norm_udot_kernel<<<nrows / 8, 256, 0, stream>>>(Ob, Ub, UD, nrows);

    out_gemm_kernel<<<dim3(D_ / 64, M_ / 64), 128, 0, stream>>>(
        UD, Wo, o_b + (size_t)i * D_, Xbuf);
  }

  ln_kernel<false><<<M_, 256, 0, stream>>>(
      Xbuf, last_ln_w, last_ln_b, (void*)out, 1e-8f);
}
